// MicroContinuumCell_18769007084085
// MI455X (gfx1250) — compile-verified, matrix operand order checked by reading, not run on hardware
//
#include <hip/hip_runtime.h>
#include <math.h>

// ============================================================================
// MicroContinuumCell fused kernel for MI455X (gfx1250), bf16 WMMA path.
//
// Roofline: ~1.0e11 FLOPs, ~1.1 GB min traffic -> BW-bound (~47us @ 23.3TB/s)
// with GEMMs on v_wmma_f32_16x16x32_bf16.  All three GEMMs (x@V^T, x@W^T,
// v^T@x) are fused into one streaming pass over x; hebb partials persist in
// registers across 8 row-tiles per block (one atomic flush per block).
//
// Register budget: 8-wave blocks -> 2 waves/SIMD -> 512 VGPRs/wave.
// hacc[2][16] (persistent hebb) = 256 VGPRs; inner loops are kept ROLLED
// (#pragma unroll 1) so loop-invariant weight fragments are re-read from
// cache instead of being hoisted+spilled to scratch.
//
// ws layout (bytes), total required ~270.5 MB:
//   [0)        Wbf   bf16 [256x256]              131072
//   [131072)   Vbf   bf16 [256x256]              131072
//   [262144)   hebb  f32  [256x256] accumulator  262144
//   [524288)   mem_t f32  [256x256]              262144
//   [786432)   colsq f32  [256]  (sum v^2 /col)    1024
//   [787456)   scal  f32  [4]: wnorm2,c0sum,c2sum,memnorm2
//   [1048576)  g0    f32  [N]   (v.gate_w + b)  1048576
//   [2097152)  v     f32  [N x 256]           268435456
// ============================================================================

#define DIM 256
#define NROWS 262144
#define TILE_M 64
#define P1_BLOCKS 512
#define TILES_PER_BLOCK ((NROWS / TILE_M) / P1_BLOCKS)  // 8

typedef __attribute__((ext_vector_type(16))) __bf16 bf16x16;
typedef __attribute__((ext_vector_type(8)))  float  f32x8;

#define OFF_WBF   ((size_t)0)
#define OFF_VBF   ((size_t)131072)
#define OFF_HEBB  ((size_t)262144)
#define OFF_MEM   ((size_t)524288)
#define OFF_COLSQ ((size_t)786432)
#define OFF_SCAL  ((size_t)787456)
#define OFF_G0    ((size_t)1048576)
#define OFF_V     ((size_t)2097152)

__device__ __forceinline__ unsigned short f2bf(float f) {
  unsigned u = __builtin_bit_cast(unsigned, f);
  u += 0x7FFFu + ((u >> 16) & 1u);            // round-to-nearest-even
  return (unsigned short)(u >> 16);
}
__device__ __forceinline__ float sigmoidf_(float x) { return 1.f / (1.f + expf(-x)); }

// ---------------------------------------------------------------------------
__global__ __launch_bounds__(256)
void zero_k(float* __restrict__ hebb, float* __restrict__ colsq, float* __restrict__ scal) {
  int idx = blockIdx.x * 256 + threadIdx.x;   // 65536 total
  hebb[idx] = 0.f;
  if (idx < DIM) colsq[idx] = 0.f;
  if (idx < 4)  scal[idx] = 0.f;
}

// Convert W/V to bf16, accumulate ||W||_F^2 into scal[0].
__global__ __launch_bounds__(256)
void prep_k(const float* __restrict__ W, const float* __restrict__ V,
            unsigned short* __restrict__ Wbf, unsigned short* __restrict__ Vbf,
            float* __restrict__ scal) {
  int idx = blockIdx.x * 256 + threadIdx.x;
  float w = W[idx], v = V[idx];
  Wbf[idx] = f2bf(w);
  Vbf[idx] = f2bf(v);
  __shared__ float red[256];
  red[threadIdx.x] = w * w;
  __syncthreads();
  for (int s = 128; s > 0; s >>= 1) {
    if ((int)threadIdx.x < s) red[threadIdx.x] += red[threadIdx.x + s];
    __syncthreads();
  }
  if (threadIdx.x == 0) atomicAdd(&scal[0], red[0]);
}

// ---------------------------------------------------------------------------
// Fused phase 1: per 64-row tile, 8 waves/block, persistent over 8 tiles.
//   wave w: main-GEMM strip = w>>1 (16 rows), j-half = (w&1)*8.
//   hebb:  wave w owns output rows [32w, 32w+32), accumulated in registers.
__global__ __launch_bounds__(256)
void phase1_k(const float* __restrict__ x,
              const unsigned short* __restrict__ Wbf,
              const unsigned short* __restrict__ Vbf,
              const float* __restrict__ gw, const float* __restrict__ gb,
              const float* __restrict__ s1w, const float* __restrict__ s1b,
              const float* __restrict__ lng, const float* __restrict__ lnb,
              const float* __restrict__ s2w, const float* __restrict__ s2b,
              const float* __restrict__ aw,  const float* __restrict__ ab,
              float* __restrict__ v_out, float* __restrict__ g0,
              float* __restrict__ hebb, float* __restrict__ colsq,
              float* __restrict__ scal)
{
  // ~100 KB static LDS (CDNA5 WGP has 320 KB)
  __shared__ __align__(16) unsigned short xbf[TILE_M][DIM];  // x tile, row-major bf16
  __shared__ __align__(16) unsigned short xT[DIM][TILE_M];   // x tile, transposed
  __shared__ __align__(16) unsigned short vT[DIM][TILE_M];   // v tile, transposed
  __shared__ float rsum[TILE_M], rsq[TILE_M], excs[TILE_M], dots[TILE_M];
  __shared__ float colsq_l[DIM];
  __shared__ float csum[2];

  const int tid   = threadIdx.x;
  const int lane  = tid & 31;
  const int wave  = tid >> 5;
  const int nl    = lane & 15;
  const int hi    = lane >> 4;
  const int strip = wave >> 1;
  const int jbase = (wave & 1) * 8;
  const int rml   = strip * 16;

  for (int i = tid; i < DIM; i += 256) colsq_l[i] = 0.f;
  if (tid < 2) csum[tid] = 0.f;

  const f32x8 vzero = {};
  f32x8 hacc[2][16];                           // persistent hebb accumulators (256 VGPRs)
  #pragma unroll
  for (int a = 0; a < 2; ++a)
    #pragma unroll
    for (int b = 0; b < 16; ++b) hacc[a][b] = vzero;

  #pragma unroll 1
  for (int tt = 0; tt < TILES_PER_BLOCK; ++tt) {
    const size_t row0 = ((size_t)blockIdx.x * TILES_PER_BLOCK + tt) * TILE_M;

    if (tid < TILE_M) { rsum[tid] = 0.f; rsq[tid] = 0.f; excs[tid] = 0.f; dots[tid] = 0.f; }
    __syncthreads();

    // --- cooperative load + fp32 row stats + bf16 convert (both layouts) ---
    const float4* xv = (const float4*)(x + row0 * DIM);
    #pragma unroll 1
    for (int it = 0; it < 16; ++it) {
      int i4 = tid + 256 * it;                 // 4096 float4 per tile
      float4 f = xv[i4];
      int o = i4 << 2, r = o >> 8, c = o & (DIM - 1);
      unsigned short b0 = f2bf(f.x), b1 = f2bf(f.y), b2 = f2bf(f.z), b3 = f2bf(f.w);
      xbf[r][c] = b0; xbf[r][c + 1] = b1; xbf[r][c + 2] = b2; xbf[r][c + 3] = b3;
      xT[c][r] = b0; xT[c + 1][r] = b1; xT[c + 2][r] = b2; xT[c + 3][r] = b3;
      atomicAdd(&rsum[r], f.x + f.y + f.z + f.w);
      atomicAdd(&rsq[r],  f.x * f.x + f.y * f.y + f.z * f.z + f.w * f.w);
    }
    if (tt + 1 < TILES_PER_BLOCK)              // hint next tile into caches
      __builtin_prefetch((const char*)(x + (row0 + TILE_M) * DIM) + tid * 256, 0, 1);
    __syncthreads();

    // --- main GEMMs: v = clamp(x@V^T), h = x@W^T ---------------------------
    float excA[8], dotA[8];
    #pragma unroll
    for (int r = 0; r < 8; ++r) { excA[r] = 0.f; dotA[r] = 0.f; }

    // jj loop kept ROLLED: B-fragment addresses vary per iteration so the
    // (tile-invariant) weight loads cannot be hoisted+spilled across tiles.
    #pragma unroll 1
    for (int jj = 0; jj < 8; ++jj) {
      const int j  = jbase + jj;
      const int jc = j * 16 + nl;
      f32x8 accV = {}; f32x8 accH = {};
      #pragma unroll
      for (int k = 0; k < 8; ++k) {
        bf16x16 a  = *(const bf16x16*)&xbf[rml + nl][k * 32 + hi * 16];
        bf16x16 bV = *(const bf16x16*)&Vbf[(size_t)jc * DIM + k * 32 + hi * 16];
        bf16x16 bW = *(const bf16x16*)&Wbf[(size_t)jc * DIM + k * 32 + hi * 16];
        accV = __builtin_amdgcn_wmma_f32_16x16x32_bf16(false, a, false, bV,
                                                       (short)0, accV, false, false);
        accH = __builtin_amdgcn_wmma_f32_16x16x32_bf16(false, a, false, bW,
                                                       (short)0, accH, false, false);
      }
      const float g = gw[jc];
      float sq = 0.f;
      #pragma unroll
      for (int r = 0; r < 8; ++r) {
        const int m = rml + r + hi * 8;        // C layout: lanes16-31 -> M+8
        float vv = fminf(fmaxf(accV[r], -2.f), 2.f);
        v_out[(row0 + m) * DIM + jc] = vv;
        vT[jc][m] = f2bf(vv);
        excA[r] += fabsf(accH[r]);
        dotA[r] += vv * g;
        sq += vv * vv;
      }
      sq += __shfl_xor(sq, 16, 32);            // combine row halves (same column)
      if (hi == 0) atomicAdd(&colsq_l[jc], sq);
    }
    #pragma unroll
    for (int r = 0; r < 8; ++r) {              // reduce over 16 columns per half
      float e = excA[r], dd = dotA[r];
      #pragma unroll
      for (int msk = 1; msk <= 8; msk <<= 1) {
        e += __shfl_xor(e, msk, 32);
        dd += __shfl_xor(dd, msk, 32);
      }
      if (nl == 0) {
        atomicAdd(&excs[rml + r + hi * 8], e);
        atomicAdd(&dots[rml + r + hi * 8], dd);
      }
    }
    __syncthreads();                           // vT / stats complete

    // --- hebb partial: v^T @ x over this tile's 64 rows (K=2x32) -----------
    // (unrolled: hacc indices must be compile-time constants)
    #pragma unroll
    for (int it2 = 0; it2 < 2; ++it2) {
      const int ib = wave * 32 + it2 * 16;
      bf16x16 aT0 = *(const bf16x16*)&vT[ib + nl][hi * 16];
      bf16x16 aT1 = *(const bf16x16*)&vT[ib + nl][32 + hi * 16];
      #pragma unroll
      for (int j = 0; j < 16; ++j) {
        bf16x16 bT0 = *(const bf16x16*)&xT[j * 16 + nl][hi * 16];
        bf16x16 bT1 = *(const bf16x16*)&xT[j * 16 + nl][32 + hi * 16];
        hacc[it2][j] = __builtin_amdgcn_wmma_f32_16x16x32_bf16(false, aT0, false, bT0,
                                                               (short)0, hacc[it2][j], false, false);
        hacc[it2][j] = __builtin_amdgcn_wmma_f32_16x16x32_bf16(false, aT1, false, bT1,
                                                               (short)0, hacc[it2][j], false, false);
      }
    }

    // --- per-row homeostatic MLP (one thread per row) ----------------------
    if (tid < TILE_M) {
      float mean = rsum[tid] * (1.f / DIM);
      float var  = rsq[tid] * (1.f / DIM) - mean * mean;
      float st[4];
      st[0] = fabsf(var - 0.5f);               // surprise
      st[1] = excs[tid] * (1.f / DIM);         // excitation
      st[2] = sqrtf(scal[0]);                  // fatigue = ||W||_F
      st[3] = 0.5f;                            // coherence
      float z[16]; float mu = 0.f;
      #pragma unroll
      for (int o = 0; o < 16; ++o) {
        float acc = s1b[o];
        #pragma unroll
        for (int i = 0; i < 4; ++i) acc += st[i] * s1w[o * 4 + i];
        z[o] = acc; mu += acc;
      }
      mu *= (1.f / 16.f);
      float vr = 0.f;
      #pragma unroll
      for (int o = 0; o < 16; ++o) { float d = z[o] - mu; vr += d * d; }
      float inv = rsqrtf(vr * (1.f / 16.f) + 1e-5f);
      #pragma unroll
      for (int o = 0; o < 16; ++o) z[o] = tanhf((z[o] - mu) * inv * lng[o] + lnb[o]);
      float z2[8];
      #pragma unroll
      for (int o = 0; o < 8; ++o) {
        float acc = s2b[o];
        #pragma unroll
        for (int i = 0; i < 16; ++i) acc += z[i] * s2w[o * 16 + i];
        z2[o] = fmaxf(acc, 0.f);
      }
      float c0 = ab[0], c2 = ab[2];
      #pragma unroll
      for (int i = 0; i < 8; ++i) { c0 += z2[i] * aw[i]; c2 += z2[i] * aw[16 + i]; }
      atomicAdd(&csum[0], sigmoidf_(c0));
      atomicAdd(&csum[1], sigmoidf_(c2));
      g0[row0 + tid] = dots[tid] + gb[0];      // gate pre-activation
    }
    __syncthreads();                           // before next tile reuses LDS
  }

  // --- flush persistent accumulators once per block ------------------------
  #pragma unroll
  for (int it2 = 0; it2 < 2; ++it2)
    #pragma unroll
    for (int j = 0; j < 16; ++j)
      #pragma unroll
      for (int r = 0; r < 8; ++r)
        atomicAdd(&hebb[(size_t)(wave * 32 + it2 * 16 + r + hi * 8) * DIM + j * 16 + nl],
                  hacc[it2][j][r]);
  atomicAdd(&colsq[tid], colsq_l[tid]);
  if (tid == 0) { atomicAdd(&scal[1], csum[0]); atomicAdd(&scal[2], csum[1]); }
}

// ---------------------------------------------------------------------------
__global__ __launch_bounds__(256)
void memA_k(const float* __restrict__ sm, const float* __restrict__ hebb,
            const float* __restrict__ colsq, float* __restrict__ scal,
            float* __restrict__ mem_tmp)
{
  int idx = blockIdx.x * 256 + threadIdx.x;    // 65536
  int i = idx >> 8;
  float hb = hebb[idx] * (1.f / NROWS);
  float fg = colsq[i] * (1.f / NROWS) * sm[idx];
  float mr = scal[1] * (1.f / NROWS);          // metabolic_rate
  float mem = sm[idx] + tanhf(hb - fg) * mr * 0.1f;
  mem_tmp[idx] = mem;
  __shared__ float red[256];
  red[threadIdx.x] = mem * mem;
  __syncthreads();
  for (int s = 128; s > 0; s >>= 1) {
    if ((int)threadIdx.x < s) red[threadIdx.x] += red[threadIdx.x + s];
    __syncthreads();
  }
  if (threadIdx.x == 0) atomicAdd(&scal[3], red[0]);
}

__global__ __launch_bounds__(256)
void memB_k(const float* __restrict__ mem_tmp, const float* __restrict__ scal,
            float* __restrict__ out_mem)
{
  int idx = blockIdx.x * 256 + threadIdx.x;
  float s = 0.5f / fmaxf(sqrtf(scal[3]), 1e-6f);
  out_mem[idx] = mem_tmp[idx] * s;
}

// output = clip(sigmoid(g0[row]) * gate_value * v, -2, 2)
__global__ __launch_bounds__(256)
void out_k(const float* __restrict__ v, const float* __restrict__ g0,
           const float* __restrict__ scal, float* __restrict__ out)
{
  size_t i4  = (size_t)blockIdx.x * 256 + threadIdx.x;   // float4 index
  size_t row = i4 >> 6;                                  // 64 float4 per row
  float gate = sigmoidf_(g0[row]) * (scal[2] * (1.f / NROWS));
  float4 vv = ((const float4*)v)[i4];
  float4 o;
  o.x = fminf(fmaxf(gate * vv.x, -2.f), 2.f);
  o.y = fminf(fmaxf(gate * vv.y, -2.f), 2.f);
  o.z = fminf(fmaxf(gate * vv.z, -2.f), 2.f);
  o.w = fminf(fmaxf(gate * vv.w, -2.f), 2.f);
  ((float4*)out)[i4] = o;
}

// ---------------------------------------------------------------------------
extern "C" void kernel_launch(void* const* d_in, const int* in_sizes, int n_in,
                              void* d_out, int out_size, void* d_ws, size_t ws_size,
                              hipStream_t stream) {
  const float* x   = (const float*)d_in[0];
  const float* W   = (const float*)d_in[1];
  const float* V   = (const float*)d_in[2];
  const float* gw  = (const float*)d_in[3];
  const float* gb  = (const float*)d_in[4];
  const float* s1w = (const float*)d_in[5];
  const float* s1b = (const float*)d_in[6];
  const float* lng = (const float*)d_in[7];
  const float* lnb = (const float*)d_in[8];
  const float* s2w = (const float*)d_in[9];
  const float* s2b = (const float*)d_in[10];
  const float* aw  = (const float*)d_in[11];
  const float* ab  = (const float*)d_in[12];
  const float* sm  = (const float*)d_in[13];

  char* ws = (char*)d_ws;
  unsigned short* Wbf = (unsigned short*)(ws + OFF_WBF);
  unsigned short* Vbf = (unsigned short*)(ws + OFF_VBF);
  float* hebb  = (float*)(ws + OFF_HEBB);
  float* mem_t = (float*)(ws + OFF_MEM);
  float* colsq = (float*)(ws + OFF_COLSQ);
  float* scal  = (float*)(ws + OFF_SCAL);
  float* g0    = (float*)(ws + OFF_G0);
  float* vbuf  = (float*)(ws + OFF_V);
  float* outp  = (float*)d_out;

  zero_k<<<DIM * DIM / 256, 256, 0, stream>>>(hebb, colsq, scal);
  prep_k<<<DIM * DIM / 256, 256, 0, stream>>>(W, V, Wbf, Vbf, scal);
  phase1_k<<<P1_BLOCKS, 256, 0, stream>>>(x, Wbf, Vbf, gw, gb, s1w, s1b, lng, lnb,
                                          s2w, s2b, aw, ab, vbuf, g0, hebb, colsq, scal);
  memA_k<<<DIM * DIM / 256, 256, 0, stream>>>(sm, hebb, colsq, scal, mem_t);
  memB_k<<<DIM * DIM / 256, 256, 0, stream>>>(mem_t, scal, outp + (size_t)NROWS * DIM);
  out_k<<<(NROWS * (DIM / 4)) / 256, 256, 0, stream>>>(vbuf, g0, scal, outp);
}
